// MyRepresentation_32873679684055
// MI455X (gfx1250) — matrix operand order repaired and checked by reading, not compiled, not run on hardware
//
#include <hip/hip_runtime.h>
#include <cstddef>

typedef _Float16 h16 __attribute__((ext_vector_type(16)));
typedef _Float16 h8v __attribute__((ext_vector_type(8)));
typedef float    f8  __attribute__((ext_vector_type(8)));

// ---------------- WMMA helpers (CDNA5 wave32, 16x16x32 f16 -> f32) ----------

__device__ inline f8 wmma_f16(h16 a, h16 b, f8 c) {
  // 8 args: (neg_a, A, neg_b, B, c_mod, C, reuse_a, reuse_b)
  return __builtin_amdgcn_wmma_f32_16x16x32_f16(false, a, false, b, (short)0, c,
                                                false, false);
}

// A fragment (16x32 f16) from f16 memory, row-major with given stride (halves).
// ISA layout: lanes 0-15 hold K {0..7,16..23}; lanes 16-31 hold K {8..15,24..31}.
__device__ inline h16 loadA_h16(const _Float16* base, int strideH) {
  int lane = threadIdx.x & 31;
  const _Float16* p = base + (lane & 15) * strideH + ((lane >> 4) << 3);
  h8v lo = *(const h8v*)p;
  h8v hi = *(const h8v*)(p + 16);
  return __builtin_shufflevector(lo, hi, 0,1,2,3,4,5,6,7,8,9,10,11,12,13,14,15);
}

// A fragment from f32 memory (converted on load).
__device__ inline h16 loadA_f32(const float* base, int ld, int kOff) {
  int lane = threadIdx.x & 31;
  const float* p = base + (lane & 15) * ld + kOff + ((lane >> 4) << 3);
  h16 a;
  #pragma unroll
  for (int e = 0; e < 8; ++e) a[e] = (_Float16)p[e];
  #pragma unroll
  for (int e = 0; e < 8; ++e) a[8 + e] = (_Float16)p[16 + e];
  return a;
}

// Gathered A fragment from LDS h state (row per lane = mp index; 0 -> zeros).
__device__ inline h16 gatherA(const _Float16* hbase, int idx, int kOff) {
  int lane = threadIdx.x & 31;
  h16 a;
  if (idx) {
    const _Float16* p = hbase + (idx - 1) * 128 + kOff + ((lane >> 4) << 3);
    h8v lo = *(const h8v*)p;
    h8v hi = *(const h8v*)(p + 16);
    a = __builtin_shufflevector(lo, hi, 0,1,2,3,4,5,6,7,8,9,10,11,12,13,14,15);
  } else {
    #pragma unroll
    for (int e = 0; e < 16; ++e) a[e] = (_Float16)0.f;
  }
  return a;
}

// B fragment: pre-swizzled, 32B per lane contiguous.
__device__ inline h16 loadB(const _Float16* frag, int kt, int nt, int NT) {
  int lane = threadIdx.x & 31;
  return *(const h16*)(frag + ((((size_t)kt * NT + nt) * 32 + lane) << 4));
}

__device__ inline float sigm(float x) { return 1.f / (1.f + __expf(-x)); }

// ---------------- Prep: swizzle row-major f32 (Ksrc x N) into B fragments ---
// Layout per lane: element e = W[kt*32 + (lane>>4)*8 + (e<8 ? e : e+8)][nt*16 + (lane&15)]
__global__ void swizzleBKernel(const float* __restrict__ W, _Float16* __restrict__ dst,
                               int Ksrc, int N, int KT) {
  int tid = blockIdx.x * blockDim.x + threadIdx.x;
  int NT = N >> 4;
  int total = KT * NT * 32;
  if (tid >= total) return;
  int lane = tid & 31;
  int tileIdx = tid >> 5;
  int nt = tileIdx % NT, kt = tileIdx / NT;
  int n = nt * 16 + (lane & 15);
  int kb = kt * 32 + ((lane >> 4) << 3);
  _Float16* d = dst + (size_t)tid * 16;
  #pragma unroll
  for (int e = 0; e < 16; ++e) {
    int kk = kb + (e < 8 ? e : e + 8);
    d[e] = (_Float16)((kk < Ksrc) ? W[(size_t)kk * N + n] : 0.f);
  }
}

// ---------------- Kernel 1: LSTM over 4096 seqs, T=10, D=13, H=128 ----------
// Block (8 waves) handles 16 sequences; wave w owns hidden tile j=w, so its
// 20 B-fragments (160 VGPRs) stay register-resident across all 10 timesteps.
// h is double-buffered in LDS (f16); c stays in one v8f per wave.
__global__ __launch_bounds__(256, 1)
void lstm1Kernel(const float* __restrict__ cond1, const float* __restrict__ cond2,
                 const _Float16* __restrict__ fWih1, const _Float16* __restrict__ fWhh1,
                 const float* __restrict__ b1, float* __restrict__ hall) {
  __shared__ _Float16 hbuf[2][16 * 128];
  __shared__ _Float16 xbuf[16 * 32];
  int lane = threadIdx.x & 31, w = threadIdx.x >> 5;   // w == hidden tile j
  int seq0 = blockIdx.x * 16;

  // Preload this wave's weights into registers (t-invariant).
  h16 bx[4], bh[4][4];
  #pragma unroll
  for (int g = 0; g < 4; ++g) bx[g] = loadB(fWih1, 0, g * 8 + w, 32);
  #pragma unroll
  for (int kt = 0; kt < 4; ++kt)
    #pragma unroll
    for (int g = 0; g < 4; ++g) bh[kt][g] = loadB(fWhh1, kt, g * 8 + w, 32);

  for (int i = threadIdx.x; i < 2048; i += 256) hbuf[0][i] = (_Float16)0.f;

  f8 cst;
  #pragma unroll
  for (int r = 0; r < 8; ++r) cst[r] = 0.f;
  int jcol = w * 16 + (lane & 15);
  float b_i = b1[jcol], b_f = b1[128 + jcol], b_g = b1[256 + jcol], b_o = b1[384 + jcol];

  for (int t = 0; t < 10; ++t) {
    __syncthreads();                       // prev-step h writes / x readers done
    // stage x_t (16 x 13 f32 -> 16 x 32 f16, zero padded)
    for (int i = threadIdx.x; i < 512; i += 256) {
      int row = i >> 5, col = i & 31;
      float v = 0.f;
      if (col < 13) {
        int sg = seq0 + row;
        v = (sg < 2048) ? cond1[(size_t)sg * 130 + t * 13 + col]
                        : cond2[(size_t)(sg - 2048) * 130 + t * 13 + col];
      }
      xbuf[i] = (_Float16)v;
    }
    __syncthreads();

    const _Float16* hr = hbuf[t & 1];
    _Float16* hwp = hbuf[(t + 1) & 1];

    f8 ai, af, ag, ao;
    #pragma unroll
    for (int r = 0; r < 8; ++r) { ai[r] = b_i; af[r] = b_f; ag[r] = b_g; ao[r] = b_o; }

    h16 ax = loadA_h16(xbuf, 32);
    ai = wmma_f16(ax, bx[0], ai);
    af = wmma_f16(ax, bx[1], af);
    ag = wmma_f16(ax, bx[2], ag);
    ao = wmma_f16(ax, bx[3], ao);
    #pragma unroll
    for (int kt = 0; kt < 4; ++kt) {
      h16 ah = loadA_h16(hr + kt * 32, 128);
      ai = wmma_f16(ah, bh[kt][0], ai);
      af = wmma_f16(ah, bh[kt][1], af);
      ag = wmma_f16(ah, bh[kt][2], ag);
      ao = wmma_f16(ah, bh[kt][3], ao);
    }
    #pragma unroll
    for (int r = 0; r < 8; ++r) {
      float iv = sigm(ai[r]), fv = sigm(af[r]), gv = tanhf(ag[r]), ov = sigm(ao[r]);
      float cv = fv * cst[r] + iv * gv;
      cst[r] = cv;
      float hv = ov * tanhf(cv);
      int mrow = r + ((lane >> 4) << 3);
      hwp[mrow * 128 + jcol] = (_Float16)hv;
      if (t == 9) hall[(size_t)(seq0 + mrow) * 128 + jcol] = hv;
    }
  }
}

// ---------------- Kernel 2: lo = 0.5*(relu(h1@Wc+bc)+relu(h2@Wc+bc)) --------
__global__ void loGemmKernel(const float* __restrict__ hall, const _Float16* __restrict__ fWc,
                             const float* __restrict__ bc, float* __restrict__ lo) {
  int lane = threadIdx.x & 31, w = threadIdx.x >> 5;
  int tile = blockIdx.x * 4 + w;          // 512 tiles: 128 M-tiles x 4 N-tiles
  int mt = tile >> 2, ntv = tile & 3;
  f8 a1, a2;
  #pragma unroll
  for (int r = 0; r < 8; ++r) { a1[r] = 0.f; a2[r] = 0.f; }
  #pragma unroll
  for (int kt = 0; kt < 4; ++kt) {
    h16 b = loadB(fWc, kt, ntv, 4);
    a1 = wmma_f16(loadA_f32(hall + (size_t)(mt * 16) * 128, 128, kt * 32), b, a1);
    a2 = wmma_f16(loadA_f32(hall + (size_t)(2048 + mt * 16) * 128, 128, kt * 32), b, a2);
  }
  int col = ntv * 16 + (lane & 15);
  float bcv = bc[col];
  #pragma unroll
  for (int r = 0; r < 8; ++r) {
    int row = mt * 16 + r + ((lane >> 4) << 3);
    float v1 = a1[r] + bcv; v1 = v1 > 0.f ? v1 : 0.f;
    float v2 = a2[r] + bcv; v2 = v2 > 0.f ? v2 : 0.f;
    lo[(size_t)row * 64 + col] = 0.5f * (v1 + v2);
  }
}

// ---------------- Kernel 3: batchnorm stats over 2048 rows of lo ------------
__global__ void bnStatsKernel(const float* __restrict__ lo, float* __restrict__ st) {
  int k = threadIdx.x;
  if (k >= 64) return;
  float s = 0.f, s2 = 0.f;
  for (int i = 0; i < 2048; ++i) { float v = lo[(size_t)i * 64 + k]; s += v; s2 += v * v; }
  float m = s / 2048.f;
  st[k] = m;
  st[64 + k] = rsqrtf(s2 / 2048.f - m * m + 1e-5f);
}

// ---------------- Kernel 4: build out (16,128,88) ---------------------------
__global__ void buildOutKernel(const float* __restrict__ ops, const float* __restrict__ ext,
                               const float* __restrict__ card, const float* __restrict__ lo,
                               const float* __restrict__ st, const float* __restrict__ g1,
                               const float* __restrict__ be1, float* __restrict__ outT) {
  int idx = blockIdx.x * blockDim.x + threadIdx.x;
  if (idx >= 16 * 128 * 88) return;
  int f = idx % 88;
  int ln = idx / 88;                      // l*128+n
  float v;
  if (f < 15)       v = ops[(size_t)ln * 15 + f];
  else if (f < 22)  v = ext[(size_t)ln * 7 + (f - 15)];
  else if (f < 24)  v = card[(size_t)ln * 2 + (f - 22)];
  else {
    int k = f - 24;
    v = (lo[(size_t)ln * 64 + k] - st[k]) * st[64 + k] * g1[k] + be1[k];
  }
  outT[idx] = v;
}

// ---------------- Kernel 5: one-step LSTM2 on out[15] (h=0) -----------------
__global__ void lstm2Kernel(const float* __restrict__ outT, const _Float16* __restrict__ fWih2,
                            const float* __restrict__ b2, float* __restrict__ h0,
                            float* __restrict__ c0) {
  __shared__ _Float16 xl[128 * 96];
  const float* x = outT + (size_t)15 * 128 * 88;
  for (int i = threadIdx.x; i < 128 * 96; i += blockDim.x) {
    int row = i / 96, col = i % 96;
    xl[i] = (_Float16)(col < 88 ? x[(size_t)row * 88 + col] : 0.f);
  }
  __syncthreads();
  int lane = threadIdx.x & 31, w = threadIdx.x >> 5;
  #pragma unroll 1
  for (int cmb = w; cmb < 64; cmb += 8) {
    int mt = cmb >> 3, j = cmb & 7;
    int jcol = j * 16 + (lane & 15);
    f8 ai, ag, ao;
    float b_i = b2[jcol], b_g = b2[256 + jcol], b_o = b2[384 + jcol];
    #pragma unroll
    for (int r = 0; r < 8; ++r) { ai[r] = b_i; ag[r] = b_g; ao[r] = b_o; }
    #pragma unroll
    for (int kt = 0; kt < 3; ++kt) {
      h16 ax = loadA_h16(xl + mt * 16 * 96 + kt * 32, 96);
      ai = wmma_f16(ax, loadB(fWih2, kt, 0 * 8 + j, 32), ai);
      ag = wmma_f16(ax, loadB(fWih2, kt, 2 * 8 + j, 32), ag);
      ao = wmma_f16(ax, loadB(fWih2, kt, 3 * 8 + j, 32), ao);
    }
    #pragma unroll
    for (int r = 0; r < 8; ++r) {
      float cv = sigm(ai[r]) * tanhf(ag[r]);   // f-gate * 0 drops out
      float hv = sigm(ao[r]) * tanhf(cv);
      int row = mt * 16 + r + ((lane >> 4) << 3);
      c0[(size_t)row * 128 + jcol] = cv;
      h0[(size_t)row * 128 + jcol] = hv;
    }
  }
}

// ---------------- Kernel 6: persistent tree scan (15 sequential levels) -----
// Only a==0 slice reaches the output -> state is (128,128). One workgroup,
// h/c double-buffered in dynamic LDS, barriers between levels.
#define TREE_FRAG_G (4 * 8 * 32 * 16)
__global__ __launch_bounds__(256, 1)
void treeKernel(const float* __restrict__ outT, const int* __restrict__ mapping,
                const float* __restrict__ Win, const float* __restrict__ binv,
                const float* __restrict__ blh, const float* __restrict__ brh,
                const _Float16* __restrict__ fWlh, const _Float16* __restrict__ fWrh,
                const float* __restrict__ h0, const float* __restrict__ c0,
                float* __restrict__ hfin) {
  extern __shared__ char smem[];
  _Float16* hb0 = (_Float16*)smem;                       // 128x128 f16
  _Float16* hb1 = hb0 + 16384;
  float* cb0 = (float*)(smem + 65536);                   // 128x128 f32
  float* cb1 = cb0 + 16384;
  float* biasg = (float*)(smem + 65536 + 131072);        // 5x128
  int* mpl = (int*)(biasg + 640);
  int* mpr = mpl + 128;

  for (int i = threadIdx.x; i < 16384; i += blockDim.x) {
    hb0[i] = (_Float16)h0[i];
    cb0[i] = c0[i];
  }
  __syncthreads();

  int lane = threadIdx.x & 31, w = threadIdx.x >> 5;
  #pragma unroll 1
  for (int s = 0; s < 15; ++s) {
    const int tIdx = 14 - s;
    if (threadIdx.x < 128) {
      int k = threadIdx.x;
      const float* lv = outT + (size_t)tIdx * 128 * 88;   // row n=0 of level
      float acc = binv[k];
      for (int f = 0; f < 88; ++f) acc += lv[f] * Win[f * 128 + k];
      #pragma unroll
      for (int g = 0; g < 5; ++g) biasg[g * 128 + k] = acc + blh[g * 128 + k] + brh[g * 128 + k];
      mpl[k] = mapping[((size_t)tIdx * 128 + k) * 2 + 0];
      mpr[k] = mapping[((size_t)tIdx * 128 + k) * 2 + 1];
    }
    __syncthreads();
    const _Float16* hc = (s & 1) ? hb1 : hb0;
    _Float16* hn = (s & 1) ? hb0 : hb1;
    const float* cc = (s & 1) ? cb1 : cb0;
    float* cn = (s & 1) ? cb0 : cb1;
    #pragma unroll 1
    for (int p = 0; p < 8; ++p) {
      int tile = w * 8 + p;
      int nt = tile >> 3, kt = tile & 7;
      int kcol = kt * 16 + (lane & 15);
      f8 acc[5];
      #pragma unroll
      for (int g = 0; g < 5; ++g) {
        float b = biasg[g * 128 + kcol];
        #pragma unroll
        for (int r = 0; r < 8; ++r) acc[g][r] = b;
      }
      int il = mpl[nt * 16 + (lane & 15)];
      int ir = mpr[nt * 16 + (lane & 15)];
      #pragma unroll 1
      for (int kk = 0; kk < 4; ++kk) {
        h16 aL = gatherA(hc, il, kk * 32);
        h16 aR = gatherA(hc, ir, kk * 32);
        #pragma unroll
        for (int g = 0; g < 5; ++g) {
          acc[g] = wmma_f16(aL, loadB(fWlh + (size_t)g * TREE_FRAG_G, kk, kt, 8), acc[g]);
          acc[g] = wmma_f16(aR, loadB(fWrh + (size_t)g * TREE_FRAG_G, kk, kt, 8), acc[g]);
        }
      }
      #pragma unroll
      for (int r = 0; r < 8; ++r) {
        int nr = nt * 16 + r + ((lane >> 4) << 3);
        int jl = mpl[nr], jr = mpr[nr];
        float lc = jl ? cc[(jl - 1) * 128 + kcol] : 0.f;
        float rc = jr ? cc[(jr - 1) * 128 + kcol] : 0.f;
        float iv = sigm(acc[0][r]), lf = sigm(acc[1][r]), rf = sigm(acc[2][r]);
        float uv = tanhf(acc[3][r]), ov = sigm(acc[4][r]);
        float cv = iv * uv + lf * lc + rf * rc;
        float hv = ov * tanhf(cv);
        cn[nr * 128 + kcol] = cv;
        hn[nr * 128 + kcol] = (_Float16)hv;
        if (s == 14) hfin[(size_t)nr * 128 + kcol] = hv;
      }
    }
    __syncthreads();
  }
}

// ---------------- Kernel 7: head (bn -> relu fc -> bn -> relu fc -> sigmoid)
__global__ void headKernel(const float* __restrict__ hfin, const float* __restrict__ g2,
                           const float* __restrict__ be2, const float* __restrict__ Wt1,
                           const float* __restrict__ bt1, const float* __restrict__ g3,
                           const float* __restrict__ be3, const float* __restrict__ Wt2,
                           const float* __restrict__ bt2, const float* __restrict__ Wo,
                           const float* __restrict__ bo, float* __restrict__ t2,
                           float* __restrict__ t3, float* __restrict__ outp) {
  __shared__ float st1[256];
  __shared__ float st2[128];
  int tid = threadIdx.x;
  if (tid < 128) {
    float s = 0.f, s2 = 0.f;
    for (int i = 0; i < 128; ++i) { float v = hfin[i * 128 + tid]; s += v; s2 += v * v; }
    float m = s / 128.f;
    st1[tid] = m; st1[128 + tid] = rsqrtf(s2 / 128.f - m * m + 1e-5f);
  }
  __syncthreads();
  for (int idx = tid; idx < 128 * 64; idx += blockDim.x) {
    int row = idx >> 6, col = idx & 63;
    float acc = bt1[col];
    for (int h = 0; h < 128; ++h) {
      float xn = (hfin[row * 128 + h] - st1[h]) * st1[128 + h] * g2[h] + be2[h];
      acc += xn * Wt1[h * 64 + col];
    }
    t2[idx] = acc > 0.f ? acc : 0.f;
  }
  __threadfence();
  __syncthreads();
  if (tid < 64) {
    float s = 0.f, s2 = 0.f;
    for (int i = 0; i < 128; ++i) { float v = t2[i * 64 + tid]; s += v; s2 += v * v; }
    float m = s / 128.f;
    st2[tid] = m; st2[64 + tid] = rsqrtf(s2 / 128.f - m * m + 1e-5f);
  }
  __syncthreads();
  for (int idx = tid; idx < 128 * 64; idx += blockDim.x) {
    int row = idx >> 6, col = idx & 63;
    float acc = bt2[col];
    for (int h = 0; h < 64; ++h) {
      float xn = (t2[row * 64 + h] - st2[h]) * st2[64 + h] * g3[h] + be3[h];
      acc += xn * Wt2[h * 64 + col];
    }
    t3[idx] = acc > 0.f ? acc : 0.f;
  }
  __threadfence();
  __syncthreads();
  if (tid < 128) {
    float acc = bo[0];
    for (int h = 0; h < 64; ++h) acc += t3[tid * 64 + h] * Wo[h];
    outp[tid] = sigm(acc);
  }
}

// ---------------- Host orchestration ----------------------------------------
extern "C" void kernel_launch(void* const* d_in, const int* in_sizes, int n_in,
                              void* d_out, int out_size, void* d_ws, size_t ws_size,
                              hipStream_t stream) {
  (void)in_sizes; (void)n_in; (void)out_size; (void)ws_size;
  const float* ops   = (const float*)d_in[0];
  const float* ext   = (const float*)d_in[1];
  const float* card  = (const float*)d_in[2];
  const float* cond1 = (const float*)d_in[3];
  const float* cond2 = (const float*)d_in[4];
  const int*   mapping = (const int*)d_in[5];
  const float* Wih1 = (const float*)d_in[6];
  const float* Whh1 = (const float*)d_in[7];
  const float* b1   = (const float*)d_in[8];
  const float* Wc   = (const float*)d_in[9];
  const float* bc   = (const float*)d_in[10];
  const float* g1   = (const float*)d_in[11];
  const float* be1  = (const float*)d_in[12];
  const float* Wih2 = (const float*)d_in[13];
  const float* b2   = (const float*)d_in[15];
  const float* Win  = (const float*)d_in[16];
  const float* binv = (const float*)d_in[17];
  const float* Wlh  = (const float*)d_in[18];
  const float* blh  = (const float*)d_in[19];
  const float* Wrh  = (const float*)d_in[20];
  const float* brh  = (const float*)d_in[21];
  const float* g2   = (const float*)d_in[22];
  const float* be2  = (const float*)d_in[23];
  const float* Wt1  = (const float*)d_in[24];
  const float* bt1  = (const float*)d_in[25];
  const float* g3   = (const float*)d_in[26];
  const float* be3  = (const float*)d_in[27];
  const float* Wt2  = (const float*)d_in[28];
  const float* bt2  = (const float*)d_in[29];
  const float* Wo   = (const float*)d_in[30];
  const float* bo   = (const float*)d_in[31];

  char* ws = (char*)d_ws;
  size_t off = 0;
  auto take = [&](size_t bytes) -> char* {
    char* r = ws + off;
    off = (off + bytes + 255) & ~(size_t)255;
    return r;
  };
  _Float16* fWhh1 = (_Float16*)take(4 * 32 * 32 * 16 * 2);   // 131072 B
  _Float16* fWih1 = (_Float16*)take(1 * 32 * 32 * 16 * 2);   // 32768 B
  _Float16* fWih2 = (_Float16*)take(3 * 32 * 32 * 16 * 2);   // 98304 B
  _Float16* fWc   = (_Float16*)take(4 * 4 * 32 * 16 * 2);    // 16384 B
  _Float16* fWlh  = (_Float16*)take(5 * TREE_FRAG_G * 2);    // 163840 B
  _Float16* fWrh  = (_Float16*)take(5 * TREE_FRAG_G * 2);    // 163840 B
  float* hall = (float*)take((size_t)4096 * 128 * 4);
  float* lo   = (float*)take((size_t)2048 * 64 * 4);
  float* st   = (float*)take(128 * 4);
  float* outT = (float*)take((size_t)16 * 128 * 88 * 4);
  float* h0   = (float*)take((size_t)128 * 128 * 4);
  float* c0   = (float*)take((size_t)128 * 128 * 4);
  float* hfin = (float*)take((size_t)128 * 128 * 4);
  float* t2   = (float*)take((size_t)128 * 64 * 4);
  float* t3   = (float*)take((size_t)128 * 64 * 4);

  // ---- weight swizzle (f32 -> f16 WMMA B fragments) ----
  swizzleBKernel<<<4, 256, 0, stream>>>(Wih1, fWih1, 13, 512, 1);
  swizzleBKernel<<<16, 256, 0, stream>>>(Whh1, fWhh1, 128, 512, 4);
  swizzleBKernel<<<12, 256, 0, stream>>>(Wih2, fWih2, 88, 512, 3);
  swizzleBKernel<<<2, 256, 0, stream>>>(Wc, fWc, 128, 64, 4);
  for (int g = 0; g < 5; ++g) {
    swizzleBKernel<<<4, 256, 0, stream>>>(Wlh + (size_t)g * 128 * 128,
                                          fWlh + (size_t)g * TREE_FRAG_G, 128, 128, 4);
    swizzleBKernel<<<4, 256, 0, stream>>>(Wrh + (size_t)g * 128 * 128,
                                          fWrh + (size_t)g * TREE_FRAG_G, 128, 128, 4);
  }

  // ---- pipeline ----
  lstm1Kernel<<<256, 256, 0, stream>>>(cond1, cond2, fWih1, fWhh1, b1, hall);
  loGemmKernel<<<128, 128, 0, stream>>>(hall, fWc, bc, lo);
  bnStatsKernel<<<1, 64, 0, stream>>>(lo, st);
  buildOutKernel<<<704, 256, 0, stream>>>(ops, ext, card, lo, st, g1, be1, outT);
  lstm2Kernel<<<1, 256, 0, stream>>>(outT, fWih2, b2, h0, c0);

  const int treeSmem = 65536 + 131072 + 640 * 4 + 256 * 4;   // 200192 B dynamic LDS
  hipFuncSetAttribute((const void*)treeKernel,
                      hipFuncAttributeMaxDynamicSharedMemorySize, treeSmem);
  treeKernel<<<1, 256, treeSmem, stream>>>(outT, mapping, Win, binv, blh, brh,
                                           fWlh, fWrh, h0, c0, hfin);

  headKernel<<<1, 256, 0, stream>>>(hfin, g2, be2, Wt1, bt1, g3, be3, Wt2, bt2,
                                    Wo, bo, t2, t3, (float*)d_out);
}
